// TrendsGRU_55087250539017
// MI455X (gfx1250) — matrix operand order repaired and checked by reading, not compiled, not run on hardware
//
#include <hip/hip_runtime.h>

typedef unsigned short u16;
typedef __attribute__((ext_vector_type(16))) __bf16 v16bf;
typedef __attribute__((ext_vector_type(8)))  float  v8f;

#define T_STEPS 512
#define BATCH   64
#define IDIM    512
#define HDIM    1024
#define ODIM    512
#define KTOT    (IDIM + HDIM)   /* 1536 */

#define NBLOCKS  32
#define NTHREADS 256
#define WAVES_PER_BLOCK (NTHREADS / 32)
#define NWAVES (NBLOCKS * WAVES_PER_BLOCK)   /* 256 wave32 = one 16x16 H-tile each */

// ---------- helpers ----------

__device__ __forceinline__ u16 f2bf(float f) {
  unsigned u = __float_as_uint(f);
  u += 0x7FFFu + ((u >> 16) & 1u);   // round-to-nearest-even
  return (u16)(u >> 16);
}

__device__ __forceinline__ float sigmoidf_fast(float x) {
  return 1.0f / (1.0f + __expf(-x));   // v_exp_f32 (TRANS)
}

union frag16 { v16bf v; uint4 q[2]; };

// Device-wide split barrier: monotonic counter, target = epoch * NBLOCKS.
// Counter is zeroed by a memset captured in the same graph -> replay-safe.
__device__ __forceinline__ void grid_sync(unsigned* cnt, unsigned target) {
  __syncthreads();
  if (threadIdx.x == 0) {
    __threadfence();
    __hip_atomic_fetch_add(cnt, 1u, __ATOMIC_RELEASE, __HIP_MEMORY_SCOPE_AGENT);
    while (__hip_atomic_load(cnt, __ATOMIC_ACQUIRE, __HIP_MEMORY_SCOPE_AGENT) < target)
      __builtin_amdgcn_s_sleep(2);
    __threadfence();
  }
  __syncthreads();
}

// ---------- WMMA tile GEMMs ----------
// A 16x32 bf16 layout: lane half h -> elems 0..15 = K+16h.. (contiguous)
// B 32x16 bf16 layout: lane reads 32 contiguous bytes of Wt[n] at K + 16h.

// Dual gate h-part: acc0 += h @ Wz_h, acc1 += h @ Wr_h (shared A fragment).
__device__ __forceinline__ void gemm2_hpart(
    const u16* __restrict__ hmat,                     // [BATCH][HDIM] bf16
    const u16* __restrict__ W0, const u16* __restrict__ W1,  // [N][KTOT]
    int mt, int nt, int lane, v8f& acc0, v8f& acc1)
{
  const int l15 = lane & 15, half = lane >> 4;
  const u16* ar = hmat + (mt * 16 + l15) * HDIM;
  const u16* w0 = W0 + (size_t)(nt * 16 + l15) * KTOT + IDIM + half * 16;
  const u16* w1 = W1 + (size_t)(nt * 16 + l15) * KTOT + IDIM + half * 16;
#pragma unroll 4
  for (int k0 = 0; k0 < HDIM; k0 += 32) {
    frag16 A, B0, B1;
    A.q[0]  = *(const uint4*)(ar + k0 + half * 16);
    A.q[1]  = *(const uint4*)(ar + k0 + half * 16 + 8);
    B0.q[0] = *(const uint4*)(w0 + k0);
    B0.q[1] = *(const uint4*)(w0 + k0 + 8);
    B1.q[0] = *(const uint4*)(w1 + k0);
    B1.q[1] = *(const uint4*)(w1 + k0 + 8);
    acc0 = __builtin_amdgcn_wmma_f32_16x16x32_bf16(false, A.v, false, B0.v,
                                                   (short)0, acc0, false, false);
    acc1 = __builtin_amdgcn_wmma_f32_16x16x32_bf16(false, A.v, false, B1.v,
                                                   (short)0, acc1, false, false);
  }
}

// Single h-part: acc += hr @ Wh_h
__device__ __forceinline__ void gemm1_hpart(
    const u16* __restrict__ hmat, const u16* __restrict__ W,
    int mt, int nt, int lane, v8f& acc)
{
  const int l15 = lane & 15, half = lane >> 4;
  const u16* ar = hmat + (mt * 16 + l15) * HDIM;
  const u16* w  = W + (size_t)(nt * 16 + l15) * KTOT + IDIM + half * 16;
#pragma unroll 4
  for (int k0 = 0; k0 < HDIM; k0 += 32) {
    frag16 A, B;
    A.q[0] = *(const uint4*)(ar + k0 + half * 16);
    A.q[1] = *(const uint4*)(ar + k0 + half * 16 + 8);
    B.q[0] = *(const uint4*)(w + k0);
    B.q[1] = *(const uint4*)(w + k0 + 8);
    acc = __builtin_amdgcn_wmma_f32_16x16x32_bf16(false, A.v, false, B.v,
                                                  (short)0, acc, false, false);
  }
}

// Triple x-part (off the critical path): az += x@Wz_x, ar += x@Wr_x, ah += x@Wh_x
__device__ __forceinline__ void gemm3_xpart(
    const u16* __restrict__ xt,                       // [BATCH][IDIM] bf16
    const u16* __restrict__ Wz, const u16* __restrict__ Wr,
    const u16* __restrict__ Wh,
    int mt, int nt, int lane, v8f& az, v8f& ar_, v8f& ah)
{
  const int l15 = lane & 15, half = lane >> 4;
  const u16* xr = xt + (mt * 16 + l15) * IDIM;
  const size_t ro = (size_t)(nt * 16 + l15) * KTOT + half * 16;
  const u16* wz = Wz + ro;
  const u16* wr = Wr + ro;
  const u16* wh = Wh + ro;
#pragma unroll 2
  for (int k0 = 0; k0 < IDIM; k0 += 32) {
    frag16 A, B0, B1, B2;
    A.q[0]  = *(const uint4*)(xr + k0 + half * 16);
    A.q[1]  = *(const uint4*)(xr + k0 + half * 16 + 8);
    B0.q[0] = *(const uint4*)(wz + k0);
    B0.q[1] = *(const uint4*)(wz + k0 + 8);
    B1.q[0] = *(const uint4*)(wr + k0);
    B1.q[1] = *(const uint4*)(wr + k0 + 8);
    B2.q[0] = *(const uint4*)(wh + k0);
    B2.q[1] = *(const uint4*)(wh + k0 + 8);
    az  = __builtin_amdgcn_wmma_f32_16x16x32_bf16(false, A.v, false, B0.v,
                                                  (short)0, az, false, false);
    ar_ = __builtin_amdgcn_wmma_f32_16x16x32_bf16(false, A.v, false, B1.v,
                                                  (short)0, ar_, false, false);
    ah  = __builtin_amdgcn_wmma_f32_16x16x32_bf16(false, A.v, false, B2.v,
                                                  (short)0, ah, false, false);
  }
}

// Output projection tile: h @ Wo  (K = HDIM, Wo transposed [ODIM][HDIM])
__device__ __forceinline__ v8f gemm_out(
    const u16* __restrict__ hmat, const u16* __restrict__ Wt,
    int mt, int nt, int lane)
{
  const int l15 = lane & 15, half = lane >> 4;
  const u16* ar = hmat + (mt * 16 + l15) * HDIM;
  const u16* w  = Wt + (size_t)(nt * 16 + l15) * HDIM + half * 16;
  v8f acc = {0.f, 0.f, 0.f, 0.f, 0.f, 0.f, 0.f, 0.f};
#pragma unroll 4
  for (int k0 = 0; k0 < HDIM; k0 += 32) {
    frag16 A, B;
    A.q[0] = *(const uint4*)(ar + k0 + half * 16);
    A.q[1] = *(const uint4*)(ar + k0 + half * 16 + 8);
    B.q[0] = *(const uint4*)(w + k0);
    B.q[1] = *(const uint4*)(w + k0 + 8);
    acc = __builtin_amdgcn_wmma_f32_16x16x32_bf16(false, A.v, false, B.v,
                                                  (short)0, acc, false, false);
  }
  return acc;
}

// ---------- conversion kernels (one-time) ----------

__global__ void transpose_bf16_k(const float* __restrict__ W, u16* __restrict__ Wt,
                                 int K, int N) {
  int total = K * N;
  for (int idx = blockIdx.x * blockDim.x + threadIdx.x; idx < total;
       idx += gridDim.x * blockDim.x) {
    int k = idx / N, n = idx % N;            // coalesced read of W[k][n]
    Wt[(size_t)n * K + k] = f2bf(W[idx]);    // Wt[n][k]
  }
}

__global__ void convert_bf16_k(const float* __restrict__ X, u16* __restrict__ Xb,
                               int total) {
  for (int idx = blockIdx.x * blockDim.x + threadIdx.x; idx < total;
       idx += gridDim.x * blockDim.x)
    Xb[idx] = f2bf(X[idx]);
}

// ---------- persistent GRU kernel ----------
// 256 waves; wave w permanently owns H-tile (mt = w&3, nt = w>>2).
// Per-wave registers carry: h (fp32, whole scan), z, and the x-part
// accumulators of the NEXT step's three gate GEMMs.
// Step: P1 = h-part of z,r (dual WMMA) | gsync | P2 = hr-part of hcand +
// h update | gsync | P3 = out projection (waves 0..127) + x-part precompute
// for step t+1 (all waves). Only 2 grid syncs per step.

__global__ __launch_bounds__(NTHREADS)
void gru_persistent(const u16* __restrict__ xbf,
                    const u16* __restrict__ Wzt, const u16* __restrict__ Wrt,
                    const u16* __restrict__ Wht, const u16* __restrict__ Wot,
                    const float* __restrict__ bz, const float* __restrict__ br,
                    const float* __restrict__ bh, const float* __restrict__ bo,
                    u16* __restrict__ hbf, u16* __restrict__ hrbf,
                    float* __restrict__ out, unsigned* __restrict__ sync_cnt)
{
  const int lane = threadIdx.x & 31;
  const int wave = blockIdx.x * WAVES_PER_BLOCK + (threadIdx.x >> 5);
  const int l15 = lane & 15, half = lane >> 4;

  // gate tile ownership (fixed for the whole scan)
  const int mt = wave & 3, nt = wave >> 2;          // nt in 0..63
  const int n  = nt * 16 + l15;
  const int mb = mt * 16 + half * 8;
  const int ebase = mb * HDIM + n;                  // epilogue index base
  const float bzv = bz[n], brv = br[n], bhv = bh[n];

  // output tile ownership (waves 0..127)
  const bool do_out = wave < 128;
  const int omt = wave & 3, ont = (wave >> 2) & 31; // nt in 0..31
  const int on  = ont * 16 + l15;
  const int omb = omt * 16 + half * 8;
  const float bov = do_out ? bo[on] : 0.0f;

  const v8f vzero = {0.f, 0.f, 0.f, 0.f, 0.f, 0.f, 0.f, 0.f};
  float hold[8];
#pragma unroll
  for (int i = 0; i < 8; ++i) hold[i] = 0.0f;       // h0 = 0 lives in VGPRs

  // x-part accumulators for step 0
  v8f az = vzero, arr = vzero, ah = vzero;
  gemm3_xpart(xbf, Wzt, Wrt, Wht, mt, nt, lane, az, arr, ah);

  unsigned epoch = 0;
  for (int t = 0; t < T_STEPS; ++t) {
    // ---- Phase 1: finish z, r with the recurrent h-part ----
    gemm2_hpart(hbf, Wzt, Wrt, mt, nt, lane, az, arr);
    float zv[8];
#pragma unroll
    for (int i = 0; i < 8; ++i) {
      zv[i] = sigmoidf_fast(az[i] + bzv);
      const float r = sigmoidf_fast(arr[i] + brv);
      hrbf[ebase + i * HDIM] = f2bf(hold[i] * r);
    }
    ++epoch; grid_sync(sync_cnt, epoch * NBLOCKS);

    // ---- Phase 2: candidate (hr-part) + h update ----
    gemm1_hpart(hrbf, Wht, mt, nt, lane, ah);
#pragma unroll
    for (int i = 0; i < 8; ++i) {
      const float hc = sigmoidf_fast(ah[i] + bhv);
      const float hn = zv[i] * hold[i] + (1.0f - zv[i]) * hc;
      hold[i] = hn;
      hbf[ebase + i * HDIM] = f2bf(hn);
    }
    ++epoch; grid_sync(sync_cnt, epoch * NBLOCKS);

    // ---- Phase 3: output projection + next step's x-parts (off crit path) ----
    if (do_out) {
      v8f ao = gemm_out(hbf, Wot, omt, ont, lane);
      float* op = out + ((size_t)t * BATCH + omb) * ODIM + on;
#pragma unroll
      for (int i = 0; i < 8; ++i)
        op[i * ODIM] = ao[i] + bov;
    }
    az = vzero; arr = vzero; ah = vzero;
    if (t + 1 < T_STEPS)
      gemm3_xpart(xbf + (size_t)(t + 1) * BATCH * IDIM,
                  Wzt, Wrt, Wht, mt, nt, lane, az, arr, ah);
  }
}

// ---------- launch ----------

extern "C" void kernel_launch(void* const* d_in, const int* in_sizes, int n_in,
                              void* d_out, int out_size, void* d_ws, size_t ws_size,
                              hipStream_t stream) {
  (void)in_sizes; (void)n_in; (void)out_size; (void)ws_size;

  const float* x  = (const float*)d_in[0];
  const float* Wz = (const float*)d_in[1];
  const float* bz = (const float*)d_in[2];
  const float* Wr = (const float*)d_in[3];
  const float* br = (const float*)d_in[4];
  const float* Wh = (const float*)d_in[5];
  const float* bh = (const float*)d_in[6];
  const float* Wo = (const float*)d_in[7];
  const float* bo = (const float*)d_in[8];
  float* out = (float*)d_out;

  char* ws = (char*)d_ws;
  unsigned* sync_cnt = (unsigned*)ws;              ws += 256;
  u16* Wzt = (u16*)ws;  ws += (size_t)KTOT * HDIM * 2;
  u16* Wrt = (u16*)ws;  ws += (size_t)KTOT * HDIM * 2;
  u16* Wht = (u16*)ws;  ws += (size_t)KTOT * HDIM * 2;
  u16* Wot = (u16*)ws;  ws += (size_t)HDIM * ODIM * 2;
  u16* xbf = (u16*)ws;  ws += (size_t)T_STEPS * BATCH * IDIM * 2;
  u16* hbf  = (u16*)ws; ws += (size_t)BATCH * HDIM * 2;
  u16* hrbf = (u16*)ws; ws += (size_t)BATCH * HDIM * 2;

  // zero the barrier counter and h0 (bf16 matrix read by P1 at t=0)
  hipMemsetAsync(sync_cnt, 0, 256, stream);
  hipMemsetAsync(hbf, 0, (size_t)BATCH * HDIM * 2, stream);

  // one-time bf16 conversion / weight transpose (all L2-resident afterwards)
  transpose_bf16_k<<<512, 256, 0, stream>>>(Wz, Wzt, KTOT, HDIM);
  transpose_bf16_k<<<512, 256, 0, stream>>>(Wr, Wrt, KTOT, HDIM);
  transpose_bf16_k<<<512, 256, 0, stream>>>(Wh, Wht, KTOT, HDIM);
  transpose_bf16_k<<<512, 256, 0, stream>>>(Wo, Wot, HDIM, ODIM);
  convert_bf16_k<<<1024, 256, 0, stream>>>(x, xbf, T_STEPS * BATCH * IDIM);

  gru_persistent<<<NBLOCKS, NTHREADS, 0, stream>>>(
      xbf, Wzt, Wrt, Wht, Wot, bz, br, bh, bo,
      hbf, hrbf, out, sync_cnt);
}
